// RMVSNet_82721070120992
// MI455X (gfx1250) — matrix operand order repaired and checked by reading, not compiled, not run on hardware
//
#include <hip/hip_runtime.h>

typedef __attribute__((ext_vector_type(16))) _Float16 v16h;
typedef __attribute__((ext_vector_type(8)))  float    v8f;

#define BV    3
#define HIMG  256
#define WIMG  320
#define DPL   128
#define FHh   64
#define FWw   80
#define FC    32
#define NPIX  (FHh*FWw)          /* 5120 */
#define W2    (FWw+2)            /* 82  halo-padded width  */
#define H2    (FHh+2)            /* 66  halo-padded height */
#define P2    (W2*H2)            /* 5412 padded pixels     */
#define KSTEPS 14                /* 9 cost k-steps + 5 hidden k-steps (448 padded K) */
#define A16_MATSZ 7168           /* 14*32*16 halves per matrix */

/* K-ordering for the GRU0 GEMM (K padded 432->448):
 *   k in [0,288):   tap = k>>5,        cost channel c = k&31
 *   k in [288,432): tap = (k-288)>>4,  hidden channel  = (k-288)&15
 *   k >= 432: zero pad (realized by loading a zero halo cell)
 * B lane layout per 32-k step: k = ks*32 + (lane>=16)*16 + e -> contiguous NHWC loads.
 * ncost/h1/rh are halo-padded (zero ring) so every tap load is unconditional.   */

/* ---------------- workspace layout (float units) ---------------- */
static constexpr size_t OFF_CAMM  = 0;                          // 36 used
static constexpr size_t OFF_STATS = 64;                         // <=24 used
static constexpr size_t OFF_FEATS = 128;
static constexpr size_t FEATS_SZ  = (size_t)BV*FC*NPIX;         // 491520
static constexpr size_t OFF_H1    = OFF_FEATS + FEATS_SZ;       // 2*[P2][16]  (padded)
static constexpr size_t OFF_H2    = OFF_H1 + 2*16*(size_t)P2;   // 2*[NPIX][4]
static constexpr size_t OFF_H3    = OFF_H2 + 2*4*(size_t)NPIX;  // 2*[NPIX][2]
static constexpr size_t OFF_RH    = OFF_H3 + 2*2*(size_t)NPIX;  // [P2][16] (padded, gru0 r*h)
static constexpr size_t ZERO_CNT  = 2*16*(size_t)P2 + 2*4*(size_t)NPIX + 2*2*(size_t)NPIX + 16*(size_t)P2;
static constexpr size_t OFF_Z     = OFF_RH + 16*(size_t)P2;     // [NPIX][16] (gru0 z)
static constexpr size_t OFF_ZS    = OFF_Z  + 16*(size_t)NPIX;   // [NPIX][4] (small z)
static constexpr size_t OFF_RS    = OFF_ZS + 4*(size_t)NPIX;    // [NPIX][4] (small r*h)
static constexpr size_t OFF_SEQ   = OFF_RS + 4*(size_t)NPIX;
static constexpr size_t OFF_A16   = OFF_SEQ + (size_t)DPL*NPIX; // 3*7168 halves = 10752 floats
static constexpr size_t OFF_BIG   = OFF_A16 + 10752;            // conv scratch aliases cost volume
static constexpr size_t NCOST_FLOATS = (size_t)DPL*P2*FC/2;     // f16 volume in float units
static constexpr size_t X0_SZ = (size_t)BV*8*HIMG*WIMG;
static constexpr size_t X1_SZ = (size_t)BV*16*(HIMG/2)*(WIMG/2);

/* ---------------- small 3x3 matrix helpers ---------------- */
__device__ __forceinline__ void inv3(const float* a, float* o) {
  float d = a[0]*(a[4]*a[8]-a[5]*a[7]) - a[1]*(a[3]*a[8]-a[5]*a[6]) + a[2]*(a[3]*a[7]-a[4]*a[6]);
  float id = 1.0f / d;
  o[0] =  (a[4]*a[8]-a[5]*a[7])*id; o[1] = -(a[1]*a[8]-a[2]*a[7])*id; o[2] =  (a[1]*a[5]-a[2]*a[4])*id;
  o[3] = -(a[3]*a[8]-a[5]*a[6])*id; o[4] =  (a[0]*a[8]-a[2]*a[6])*id; o[5] = -(a[0]*a[5]-a[2]*a[3])*id;
  o[6] =  (a[3]*a[7]-a[4]*a[6])*id; o[7] = -(a[0]*a[7]-a[1]*a[6])*id; o[8] =  (a[0]*a[4]-a[1]*a[3])*id;
}
__device__ __forceinline__ void mm3(const float* a, const float* b, float* o) {
  for (int i = 0; i < 3; ++i)
    for (int j = 0; j < 3; ++j)
      o[i*3+j] = a[i*3+0]*b[0*3+j] + a[i*3+1]*b[1*3+j] + a[i*3+2]*b[2*3+j];
}
__device__ __forceinline__ void mv3(const float* a, const float* v, float* o) {
  for (int i = 0; i < 3; ++i) o[i] = a[i*3+0]*v[0] + a[i*3+1]*v[1] + a[i*3+2]*v[2];
}

/* Fold Kv*Rv*R0^-1*K0^-1 and Kv*(tv - Rv*R0^-1*t0) into 12 floats per view. */
__global__ void cam_prep(const float* __restrict__ cam, float* __restrict__ M) {
  if (threadIdx.x != 0 || blockIdx.x != 0) return;
  float R0[9], t0[3], K0[9], R0i[9], K0i[9];
  for (int i = 0; i < 3; ++i) {
    for (int j = 0; j < 3; ++j) { R0[i*3+j] = cam[i*4+j]; K0[i*3+j] = cam[16+i*4+j]; }
    t0[i] = cam[i*4+3];
  }
  inv3(R0, R0i); inv3(K0, K0i);
  for (int v = 0; v < BV; ++v) {
    const float* cv = cam + v*32;
    float Rv[9], tv[3], Kv[9];
    for (int i = 0; i < 3; ++i) {
      for (int j = 0; j < 3; ++j) { Rv[i*3+j] = cv[i*4+j]; Kv[i*3+j] = cv[16+i*4+j]; }
      tv[i] = cv[i*4+3];
    }
    float T[9], KT[9], Mv[9], Tt0[3], tmp[3], bv[3];
    mm3(Rv, R0i, T); mm3(Kv, T, KT); mm3(KT, K0i, Mv);
    mv3(T, t0, Tt0);
    tmp[0] = tv[0]-Tt0[0]; tmp[1] = tv[1]-Tt0[1]; tmp[2] = tv[2]-Tt0[2];
    mv3(Kv, tmp, bv);
    for (int i = 0; i < 9; ++i) M[v*12+i] = Mv[i];
    for (int i = 0; i < 3; ++i) M[v*12+9+i] = bv[i];
  }
}

/* ---------------- generic 3x3 conv (pad 1), CHW ---------------- */
__global__ void conv3x3(const float* __restrict__ in, float* __restrict__ out,
                        const float* __restrict__ w, const float* __restrict__ bias,
                        int nimg, int Ci, int Hi, int Wi, int Co, int Ho, int Wo, int stride) {
  int gid = blockIdx.x * blockDim.x + threadIdx.x;
  int per = Co * Ho * Wo;
  if (gid >= nimg * per) return;
  int img = gid / per, rem = gid - img * per;
  int co = rem / (Ho * Wo), p = rem - co * (Ho * Wo);
  int oy = p / Wo, ox = p - oy * Wo;
  float acc = bias ? bias[co] : 0.0f;
  for (int ci = 0; ci < Ci; ++ci)
    for (int ky = 0; ky < 3; ++ky) {
      int iy = oy * stride + ky - 1;
      if (iy < 0 || iy >= Hi) continue;
      for (int kx = 0; kx < 3; ++kx) {
        int ix = ox * stride + kx - 1;
        if (ix < 0 || ix >= Wi) continue;
        acc += in[((size_t)(img*Ci + ci)*Hi + iy)*Wi + ix] * w[((co*Ci + ci)*3 + ky)*3 + kx];
      }
    }
  out[gid] = acc;
}

/* ---------------- group norm ---------------- */
__global__ void gn_reduce(const float* __restrict__ x, float* __restrict__ stats,
                          int C, int HW, int groups) {
  __shared__ float ss[256], ss2[256];
  int img = blockIdx.x / groups, g = blockIdx.x % groups;
  int cpg = C / groups;
  size_t n = (size_t)cpg * HW;
  size_t base = ((size_t)img * C + (size_t)g * cpg) * HW;
  float s = 0.0f, s2 = 0.0f;
  for (size_t i = threadIdx.x; i < n; i += blockDim.x) {
    float v = x[base + i]; s += v; s2 += v * v;
  }
  ss[threadIdx.x] = s; ss2[threadIdx.x] = s2;
  __syncthreads();
  for (int st = 128; st > 0; st >>= 1) {
    if (threadIdx.x < st) { ss[threadIdx.x] += ss[threadIdx.x+st]; ss2[threadIdx.x] += ss2[threadIdx.x+st]; }
    __syncthreads();
  }
  if (threadIdx.x == 0) {
    float mean = ss[0] / (float)n;
    float var = ss2[0] / (float)n - mean * mean;
    stats[2*blockIdx.x]   = mean;
    stats[2*blockIdx.x+1] = rsqrtf(var + 1e-5f);
  }
}

__global__ void gn_apply(float* __restrict__ x, const float* __restrict__ stats,
                         const float* __restrict__ gamma, const float* __restrict__ beta,
                         int C, int HW, int groups, int ntot) {
  int gid = blockIdx.x * blockDim.x + threadIdx.x;
  if (gid >= ntot) return;
  int img = gid / (C * HW);
  int c = (gid / HW) % C;
  int g = c / (C / groups);
  float m = stats[2*(img*groups+g)], rs = stats[2*(img*groups+g)+1];
  float v = (x[gid] - m) * rs * gamma[c] + beta[c];
  x[gid] = fmaxf(v, 0.0f);
}

/* ---------------- plane-sweep variance cost volume (negated, f16, NHWC, halo-padded) ------- */
__global__ void cost_kernel(const float* __restrict__ feats, const float* __restrict__ M,
                            const float* __restrict__ cam, _Float16* __restrict__ ncost) {
  int gid = blockIdx.x * blockDim.x + threadIdx.x;
  if (gid >= DPL * NPIX) return;
  int d = gid / NPIX, pix = gid - d * NPIX;
  int py = pix / FWw, px = pix - py * FWw;
  float ds = cam[28], de = cam[31];
  float dep = ds + (de - ds) * (float)d / (float)(DPL - 1);
  float gx = ((float)px + 0.5f) * dep, gy = ((float)py + 0.5f) * dep, gz = dep;
  float wgt[2][4]; int off[2][4];
#pragma unroll
  for (int vi = 0; vi < 2; ++vi) {
    const float* Mv = M + (vi + 1) * 12;
    float u  = Mv[0]*gx + Mv[1]*gy + Mv[2]*gz + Mv[9];
    float vv = Mv[3]*gx + Mv[4]*gy + Mv[5]*gz + Mv[10];
    float z  = Mv[6]*gx + Mv[7]*gy + Mv[8]*gz + Mv[11];
    float iz = 1.0f / (z + 1e-9f);
    float ix = u * iz - 0.5f, iy = vv * iz - 0.5f;
    float x0 = floorf(ix), y0 = floorf(iy);
#pragma unroll
    for (int j = 0; j < 4; ++j) {
      float xc = x0 + (float)(j & 1), yc = y0 + (float)(j >> 1);
      bool inb = (xc >= 0.0f) && (xc <= (float)(FWw-1)) && (yc >= 0.0f) && (yc <= (float)(FHh-1));
      float w = (1.0f - fabsf(ix - xc)) * (1.0f - fabsf(iy - yc));
      int xi = (int)fminf(fmaxf(xc, 0.0f), (float)(FWw-1));
      int yi = (int)fminf(fmaxf(yc, 0.0f), (float)(FHh-1));
      wgt[vi][j] = inb ? w : 0.0f;
      off[vi][j] = yi * FWw + xi;
    }
  }
  _Float16* outp = ncost + ((size_t)d * P2 + (size_t)(py + 1) * W2 + (px + 1)) * FC;
  for (int c = 0; c < FC; ++c) {
    float f0 = feats[(size_t)c * NPIX + pix];   // view 0 uses ref features directly
    float s = f0, s2 = f0 * f0;
#pragma unroll
    for (int vi = 0; vi < 2; ++vi) {
      const float* fv = feats + ((size_t)(vi + 1) * FC + c) * NPIX;
      float f = wgt[vi][0]*fv[off[vi][0]] + wgt[vi][1]*fv[off[vi][1]]
              + wgt[vi][2]*fv[off[vi][2]] + wgt[vi][3]*fv[off[vi][3]];
      s += f; s2 += f * f;
    }
    float mean = s * (1.0f/3.0f);
    float cv = s2 * (1.0f/3.0f) - mean * mean;
    outp[c] = (_Float16)(-cv);                  // pre-negate: GRU input is -cost
  }
}

/* ---------------- repack GRU0 weights into CDNA5 16x32 f16 A-operand layout ---------------- */
__global__ void gru0_prep(const float* __restrict__ wz, const float* __restrict__ wr,
                          const float* __restrict__ wc, _Float16* __restrict__ a16) {
  int gid = blockIdx.x * blockDim.x + threadIdx.x;
  if (gid >= 3 * KSTEPS * 32) return;
  int mat = gid / (KSTEPS * 32);
  int rem = gid - mat * (KSTEPS * 32);
  int ks = rem / 32, lane = rem & 31;
  int m = lane & 15, hi = lane >> 4;
  const float* w = (mat == 0) ? wz : (mat == 1) ? wr : wc;
#pragma unroll
  for (int e = 0; e < 16; ++e) {
    // A 16x32 f16 lane layout: k_in_step = 16*hi + e  (contiguous, matches B)
    int k = ks * 32 + hi * 16 + e;
    float val = 0.0f;
    int tap = 0, c = -1;
    if (k < 288)      { tap = k >> 5;          c = k & 31; }              // cost part
    else if (k < 432) { tap = (k - 288) >> 4;  c = 32 + ((k-288) & 15); } // hidden part
    if (c >= 0) val = w[((m * 48 + c) * 3 + (tap / 3)) * 3 + (tap % 3)];
    a16[(size_t)gid * 16 + e] = (_Float16)val;
  }
}

/* ---------------- GRU0 z/r gates: two WMMA chains sharing one im2col B ----------------
 * ncost_d : [P2][32] f16 padded (negated cost)   h1 : [P2][16] f32 padded (zero ring)
 * outputs : zb [NPIX][16] = sigmoid(z), rh [P2][16] = sigmoid(r)*h1 (interior only)  */
__global__ __launch_bounds__(128) void gru0_zr(
    const _Float16* __restrict__ ncost_d, const float* __restrict__ h1,
    const _Float16* __restrict__ a16,
    const float* __restrict__ bz, const float* __restrict__ br,
    float* __restrict__ zb, float* __restrict__ rh) {
  const int lane = threadIdx.x & 31;
  const int tile = blockIdx.x * 4 + (threadIdx.x >> 5);
  const int hi = lane >> 4;
  const int pix = tile * 16 + (lane & 15);
  const int py = pix / FWw, px = pix - py * FWw;
  const int ppad = (py + 1) * W2 + (px + 1);

  const v16h* Az = (const v16h*)a16;
  const v16h* Ar = (const v16h*)(a16 + A16_MATSZ);
  v8f accZ = {}, accR = {};

  /* cost part: tap = ks, one unconditional contiguous 32B v16h load per k-step */
#pragma unroll
  for (int ks = 0; ks < 9; ++ks) {
    const int dy = ks / 3 - 1, dx = ks % 3 - 1;
    const v16h b = *(const v16h*)(ncost_d + (size_t)(ppad + dy * W2 + dx) * FC + hi * 16);
    v16h az = Az[ks * 32 + lane];
    v16h ar = Ar[ks * 32 + lane];
    accZ = __builtin_amdgcn_wmma_f32_16x16x32_f16(false, az, false, b, (short)0, accZ, false, false);
    accR = __builtin_amdgcn_wmma_f32_16x16x32_f16(false, ar, false, b, (short)0, accR, false, false);
  }
  /* hidden part: two taps per k-step (one per lane half); pad tap 9 -> zero halo cell 0 */
#pragma unroll
  for (int ks = 9; ks < KSTEPS; ++ks) {
    const int tap = 2 * (ks - 9) + hi;
    const int off = (tap < 9) ? (ppad + (tap / 3 - 1) * W2 + (tap % 3 - 1)) : 0;
    const float* src = h1 + (size_t)off * 16;
    v16h b;
#pragma unroll
    for (int e = 0; e < 16; ++e) b[e] = (_Float16)src[e];
    v16h az = Az[ks * 32 + lane];
    v16h ar = Ar[ks * 32 + lane];
    accZ = __builtin_amdgcn_wmma_f32_16x16x32_f16(false, az, false, b, (short)0, accZ, false, false);
    accR = __builtin_amdgcn_wmma_f32_16x16x32_f16(false, ar, false, b, (short)0, accR, false, false);
  }
#pragma unroll
  for (int i = 0; i < 8; ++i) {
    int m = hi * 8 + i;
    float z = 1.0f / (1.0f + expf(-(accZ[i] + bz[m])));
    float r = 1.0f / (1.0f + expf(-(accR[i] + br[m])));
    zb[pix * 16 + m] = z;
    rh[(size_t)ppad * 16 + m] = r * h1[(size_t)ppad * 16 + m];
  }
}

/* ---------------- GRU0 candidate conv + hidden-state update ---------------- */
__global__ __launch_bounds__(128) void gru0_cg(
    const _Float16* __restrict__ ncost_d, const float* __restrict__ h1,
    const _Float16* __restrict__ a16c, const float* __restrict__ bc,
    const float* __restrict__ zb, const float* __restrict__ rh,
    float* __restrict__ h1out) {
  const int lane = threadIdx.x & 31;
  const int tile = blockIdx.x * 4 + (threadIdx.x >> 5);
  const int hi = lane >> 4;
  const int pix = tile * 16 + (lane & 15);
  const int py = pix / FWw, px = pix - py * FWw;
  const int ppad = (py + 1) * W2 + (px + 1);

  const v16h* Ac = (const v16h*)a16c;
  v8f acc = {};
#pragma unroll
  for (int ks = 0; ks < 9; ++ks) {
    const int dy = ks / 3 - 1, dx = ks % 3 - 1;
    const v16h b = *(const v16h*)(ncost_d + (size_t)(ppad + dy * W2 + dx) * FC + hi * 16);
    v16h ac = Ac[ks * 32 + lane];
    acc = __builtin_amdgcn_wmma_f32_16x16x32_f16(false, ac, false, b, (short)0, acc, false, false);
  }
#pragma unroll
  for (int ks = 9; ks < KSTEPS; ++ks) {
    const int tap = 2 * (ks - 9) + hi;
    const int off = (tap < 9) ? (ppad + (tap / 3 - 1) * W2 + (tap % 3 - 1)) : 0;
    const float* src = rh + (size_t)off * 16;
    v16h b;
#pragma unroll
    for (int e = 0; e < 16; ++e) b[e] = (_Float16)src[e];
    v16h ac = Ac[ks * 32 + lane];
    acc = __builtin_amdgcn_wmma_f32_16x16x32_f16(false, ac, false, b, (short)0, acc, false, false);
  }
#pragma unroll
  for (int i = 0; i < 8; ++i) {
    int m = hi * 8 + i;
    float cg = tanhf(acc[i] + bc[m]);
    float z = zb[pix * 16 + m];
    float ho = h1[(size_t)ppad * 16 + m];
    h1out[(size_t)ppad * 16 + m] = z * ho + (1.0f - z) * cg;
  }
}

/* ---------------- small conv-GRU layers (<=4 out channels), NHWC ----------------
 * x: [P2][Cx] if xpad else [NPIX][Cx];  h: [NPIX][Ch] (Ch==Co).                  */
__global__ void small_zr(const float* __restrict__ x, int Cx, int xpad,
                         const float* __restrict__ h, int Ch,
                         const float* __restrict__ wz, const float* __restrict__ bz,
                         const float* __restrict__ wr, const float* __restrict__ br,
                         int Co, float* __restrict__ zb, float* __restrict__ rh) {
  int pix = blockIdx.x * blockDim.x + threadIdx.x;
  if (pix >= NPIX) return;
  int py = pix / FWw, px = pix - py * FWw;
  int Ci = Cx + Ch;
  float az[4], ar[4];
#pragma unroll
  for (int o = 0; o < 4; ++o) { az[o] = (o < Co) ? bz[o] : 0.0f; ar[o] = (o < Co) ? br[o] : 0.0f; }
  for (int ky = 0; ky < 3; ++ky) {
    int sy = py + ky - 1;
    if (sy < 0 || sy >= FHh) continue;
    for (int kx = 0; kx < 3; ++kx) {
      int sx = px + kx - 1;
      if (sx < 0 || sx >= FWw) continue;
      int sp = sy * FWw + sx;
      int spx = xpad ? ((sy + 1) * W2 + (sx + 1)) : sp;
      for (int c = 0; c < Ci; ++c) {
        float v = (c < Cx) ? x[(size_t)spx * Cx + c] : h[(size_t)sp * Ch + (c - Cx)];
#pragma unroll
        for (int o = 0; o < 4; ++o)
          if (o < Co) {
            az[o] += v * wz[((o * Ci + c) * 3 + ky) * 3 + kx];
            ar[o] += v * wr[((o * Ci + c) * 3 + ky) * 3 + kx];
          }
      }
    }
  }
#pragma unroll
  for (int o = 0; o < 4; ++o)
    if (o < Co) {
      float z = 1.0f / (1.0f + expf(-az[o]));
      float r = 1.0f / (1.0f + expf(-ar[o]));
      zb[(size_t)pix * Co + o] = z;
      rh[(size_t)pix * Co + o] = r * h[(size_t)pix * Ch + o];
    }
}

__global__ void small_cg(const float* __restrict__ x, int Cx, int xpad,
                         const float* __restrict__ h, int Ch,
                         const float* __restrict__ rh, const float* __restrict__ wc,
                         const float* __restrict__ bc, const float* __restrict__ zb,
                         int Co, float* __restrict__ hout) {
  int pix = blockIdx.x * blockDim.x + threadIdx.x;
  if (pix >= NPIX) return;
  int py = pix / FWw, px = pix - py * FWw;
  int Ci = Cx + Ch;
  float ac[4];
#pragma unroll
  for (int o = 0; o < 4; ++o) ac[o] = (o < Co) ? bc[o] : 0.0f;
  for (int ky = 0; ky < 3; ++ky) {
    int sy = py + ky - 1;
    if (sy < 0 || sy >= FHh) continue;
    for (int kx = 0; kx < 3; ++kx) {
      int sx = px + kx - 1;
      if (sx < 0 || sx >= FWw) continue;
      int sp = sy * FWw + sx;
      int spx = xpad ? ((sy + 1) * W2 + (sx + 1)) : sp;
      for (int c = 0; c < Ci; ++c) {
        float v = (c < Cx) ? x[(size_t)spx * Cx + c] : rh[(size_t)sp * Co + (c - Cx)];
#pragma unroll
        for (int o = 0; o < 4; ++o)
          if (o < Co) ac[o] += v * wc[((o * Ci + c) * 3 + ky) * 3 + kx];
      }
    }
  }
#pragma unroll
  for (int o = 0; o < 4; ++o)
    if (o < Co) {
      float cg = tanhf(ac[o]);
      float z = zb[(size_t)pix * Co + o];
      hout[(size_t)pix * Co + o] = z * h[(size_t)pix * Co + o] + (1.0f - z) * cg;
    }
}

__global__ void prob_conv(const float* __restrict__ h3, const float* __restrict__ w,
                          const float* __restrict__ b, float* __restrict__ seq_d) {
  int pix = blockIdx.x * blockDim.x + threadIdx.x;
  if (pix >= NPIX) return;
  int py = pix / FWw, px = pix - py * FWw;
  float acc = b[0];
  for (int ky = 0; ky < 3; ++ky) {
    int sy = py + ky - 1;
    if (sy < 0 || sy >= FHh) continue;
    for (int kx = 0; kx < 3; ++kx) {
      int sx = px + kx - 1;
      if (sx < 0 || sx >= FWw) continue;
      int sp = sy * FWw + sx;
      acc += h3[(size_t)sp * 2 + 0] * w[(0 * 3 + ky) * 3 + kx];
      acc += h3[(size_t)sp * 2 + 1] * w[(1 * 3 + ky) * 3 + kx];
    }
  }
  seq_d[pix] = acc;
}

/* ---------------- softmax over depth, expected depth + 4-tap prob map ---------------- */
__global__ void depth_out(const float* __restrict__ seq, const float* __restrict__ cam,
                          float* __restrict__ out) {
  int pix = blockIdx.x * blockDim.x + threadIdx.x;
  if (pix >= NPIX) return;
  float ds = cam[28], di = cam[29], de = cam[31];
  float mx = -1e30f;
  for (int d = 0; d < DPL; ++d) mx = fmaxf(mx, seq[(size_t)d * NPIX + pix]);
  float sum = 0.0f, wsum = 0.0f;
  for (int d = 0; d < DPL; ++d) {
    float e = expf(seq[(size_t)d * NPIX + pix] - mx);
    sum += e;
    wsum += e * (ds + (de - ds) * (float)d / (float)(DPL - 1));
  }
  float pred = wsum / sum;
  out[pix] = pred;
  int i0 = (int)floorf((pred - ds) / di);
  float pm = 0.0f;
#pragma unroll
  for (int j = 0; j < 4; ++j) {
    int gi = i0 + j - 1;
    gi = gi < 0 ? 0 : (gi > DPL - 1 ? DPL - 1 : gi);
    pm += expf(seq[(size_t)gi * NPIX + pix] - mx);
  }
  out[NPIX + pix] = pm / sum;
}

__global__ void zero_kernel(float* __restrict__ p, size_t n) {
  size_t g = (size_t)blockIdx.x * blockDim.x + threadIdx.x;
  if (g < n) p[g] = 0.0f;
}

/* ======================================================================= */
extern "C" void kernel_launch(void* const* d_in, const int* in_sizes, int n_in,
                              void* d_out, int out_size, void* d_ws, size_t ws_size,
                              hipStream_t stream) {
  (void)in_sizes; (void)n_in; (void)out_size; (void)ws_size;
  const float* img   = (const float*)d_in[0];
  const float* cam   = (const float*)d_in[1];
  const float* fe_w0 = (const float*)d_in[2];
  const float* fe_g0 = (const float*)d_in[3];
  const float* fe_b0 = (const float*)d_in[4];
  const float* fe_w1 = (const float*)d_in[5];
  const float* fe_g1 = (const float*)d_in[6];
  const float* fe_b1 = (const float*)d_in[7];
  const float* fe_w2 = (const float*)d_in[8];
  const float* fe_g2 = (const float*)d_in[9];
  const float* fe_b2 = (const float*)d_in[10];
  const float* fe_w3 = (const float*)d_in[11];
  const float* fe_b3 = (const float*)d_in[12];
  const float* g0wz = (const float*)d_in[13]; const float* g0bz = (const float*)d_in[14];
  const float* g0wr = (const float*)d_in[15]; const float* g0br = (const float*)d_in[16];
  const float* g0wc = (const float*)d_in[17]; const float* g0bc = (const float*)d_in[18];
  const float* g1wz = (const float*)d_in[19]; const float* g1bz = (const float*)d_in[20];
  const float* g1wr = (const float*)d_in[21]; const float* g1br = (const float*)d_in[22];
  const float* g1wc = (const float*)d_in[23]; const float* g1bc = (const float*)d_in[24];
  const float* g2wz = (const float*)d_in[25]; const float* g2bz = (const float*)d_in[26];
  const float* g2wr = (const float*)d_in[27]; const float* g2br = (const float*)d_in[28];
  const float* g2wc = (const float*)d_in[29]; const float* g2bc = (const float*)d_in[30];
  const float* pw   = (const float*)d_in[31]; const float* pb   = (const float*)d_in[32];

  float* ws = (float*)d_ws;
  float* camM  = ws + OFF_CAMM;
  float* stats = ws + OFF_STATS;
  float* feats = ws + OFF_FEATS;
  float* h1    = ws + OFF_H1;    // padded [2][P2][16]
  float* h2    = ws + OFF_H2;
  float* h3    = ws + OFF_H3;
  float* rh    = ws + OFF_RH;    // padded [P2][16]
  float* zb    = ws + OFF_Z;
  float* zs    = ws + OFF_ZS;
  float* rs    = ws + OFF_RS;
  float* seq   = ws + OFF_SEQ;
  _Float16* a16   = (_Float16*)(ws + OFF_A16);
  _Float16* ncost = (_Float16*)(ws + OFF_BIG);  // written only after conv scratch consumed
  float* x0 = ws + OFF_BIG;
  float* x1 = x0 + X0_SZ;
  float* x2 = x1 + X1_SZ;

  cam_prep<<<1, 1, 0, stream>>>(cam, camM);

  /* feature extraction pyramid */
  int n0 = BV * 8 * HIMG * WIMG;
  conv3x3<<<(n0+255)/256, 256, 0, stream>>>(img, x0, fe_w0, nullptr, BV, 3, HIMG, WIMG, 8, HIMG, WIMG, 1);
  gn_reduce<<<BV*1, 256, 0, stream>>>(x0, stats, 8, HIMG*WIMG, 1);
  gn_apply<<<(n0+255)/256, 256, 0, stream>>>(x0, stats, fe_g0, fe_b0, 8, HIMG*WIMG, 1, n0);

  int n1 = BV * 16 * (HIMG/2) * (WIMG/2);
  conv3x3<<<(n1+255)/256, 256, 0, stream>>>(x0, x1, fe_w1, nullptr, BV, 8, HIMG, WIMG, 16, HIMG/2, WIMG/2, 2);
  gn_reduce<<<BV*2, 256, 0, stream>>>(x1, stats, 16, (HIMG/2)*(WIMG/2), 2);
  gn_apply<<<(n1+255)/256, 256, 0, stream>>>(x1, stats, fe_g1, fe_b1, 16, (HIMG/2)*(WIMG/2), 2, n1);

  int n2 = BV * 32 * FHh * FWw;
  conv3x3<<<(n2+255)/256, 256, 0, stream>>>(x1, x2, fe_w2, nullptr, BV, 16, HIMG/2, WIMG/2, 32, FHh, FWw, 2);
  gn_reduce<<<BV*4, 256, 0, stream>>>(x2, stats, 32, NPIX, 4);
  gn_apply<<<(n2+255)/256, 256, 0, stream>>>(x2, stats, fe_g2, fe_b2, 32, NPIX, 4, n2);

  conv3x3<<<(n2+255)/256, 256, 0, stream>>>(x2, feats, fe_w3, fe_b3, BV, 32, FHh, FWw, 32, FHh, FWw, 1);

  /* zero padded cost volume (scratch consumed), then fill interiors */
  zero_kernel<<<(unsigned)((NCOST_FLOATS + 255) / 256), 256, 0, stream>>>(ws + OFF_BIG, NCOST_FLOATS);
  cost_kernel<<<(DPL*NPIX+255)/256, 256, 0, stream>>>(feats, camM, cam, ncost);
  gru0_prep<<<(3*KSTEPS*32+255)/256, 256, 0, stream>>>(g0wz, g0wr, g0wc, a16);
  /* zero h1/h2/h3/rh (contiguous): establishes zero halo rings + zero initial state */
  zero_kernel<<<(unsigned)((ZERO_CNT + 255) / 256), 256, 0, stream>>>(h1, ZERO_CNT);

  /* sequential GRU scan over depth */
  for (int d = 0; d < DPL; ++d) {
    const _Float16* ncost_d = ncost + (size_t)d * P2 * FC;
    float* h1i = h1 + (size_t)(d & 1) * 16 * P2;
    float* h1o = h1 + (size_t)((d + 1) & 1) * 16 * P2;
    float* h2i = h2 + (size_t)(d & 1) * 4 * NPIX;
    float* h2o = h2 + (size_t)((d + 1) & 1) * 4 * NPIX;
    float* h3i = h3 + (size_t)(d & 1) * 2 * NPIX;
    float* h3o = h3 + (size_t)((d + 1) & 1) * 2 * NPIX;

    gru0_zr<<<80, 128, 0, stream>>>(ncost_d, h1i, a16, g0bz, g0br, zb, rh);
    gru0_cg<<<80, 128, 0, stream>>>(ncost_d, h1i, a16 + 2*A16_MATSZ, g0bc, zb, rh, h1o);
    small_zr<<<20, 256, 0, stream>>>(h1o, 16, 1, h2i, 4, g1wz, g1bz, g1wr, g1br, 4, zs, rs);
    small_cg<<<20, 256, 0, stream>>>(h1o, 16, 1, h2i, 4, rs, g1wc, g1bc, zs, 4, h2o);
    small_zr<<<20, 256, 0, stream>>>(h2o, 4, 0, h3i, 2, g2wz, g2bz, g2wr, g2br, 2, zs, rs);
    small_cg<<<20, 256, 0, stream>>>(h2o, 4, 0, h3i, 2, rs, g2wc, g2bc, zs, 2, h3o);
    prob_conv<<<20, 256, 0, stream>>>(h3o, pw, pb, seq + (size_t)d * NPIX);
  }

  depth_out<<<20, 256, 0, stream>>>(seq, cam, (float*)d_out);
}
